// SAE_71236327571697
// MI455X (gfx1250) — compile-verified, matrix operand order checked by reading, not run on hardware
//
#include <hip/hip_runtime.h>
#include <hip/hip_bf16.h>
#include <stdint.h>
#include <math.h>

// Problem sizes (fixed by the reference)
#define B_SZ   8192
#define D_IN   768
#define D_HID  16384
#define K_TOP  32

typedef __attribute__((ext_vector_type(16))) __bf16 v16bf;
typedef __attribute__((ext_vector_type(8)))  __bf16 v8bf;
typedef __attribute__((ext_vector_type(4)))  __bf16 v4bf;
typedef __attribute__((ext_vector_type(8)))  float  v8f;

// ---------------------------------------------------------------------------
// Kernel 0: f32 -> bf16 pre-conversion (one float4 -> one v4bf per thread)
// ---------------------------------------------------------------------------
__global__ __launch_bounds__(256)
void f32_to_bf16(const float* __restrict__ in, __bf16* __restrict__ out)
{
    const int i = blockIdx.x * 256 + threadIdx.x;
    const float4 f = ((const float4*)in)[i];
    v4bf o;
    o[0] = (__bf16)f.x; o[1] = (__bf16)f.y; o[2] = (__bf16)f.z; o[3] = (__bf16)f.w;
    ((v4bf*)out)[i] = o;
}

// ---------------------------------------------------------------------------
// Kernel 1: transpose W_dec [D_IN, D_HID] -> W_decT [D_HID, D_IN]
// ---------------------------------------------------------------------------
__global__ __launch_bounds__(256)
void sae_transpose(const float* __restrict__ Wdec, float* __restrict__ WdecT)
{
    __shared__ float tile[32][33];
    const int h0 = blockIdx.x * 32;
    const int i0 = blockIdx.y * 32;
    const int tx = threadIdx.x & 31;
    const int ty = threadIdx.x >> 5;
#pragma unroll
    for (int r = 0; r < 4; ++r)
        tile[ty + 8 * r][tx] = Wdec[(size_t)(i0 + ty + 8 * r) * D_HID + h0 + tx];
    __syncthreads();
#pragma unroll
    for (int r = 0; r < 4; ++r)
        WdecT[(size_t)(h0 + ty + 8 * r) * D_IN + i0 + tx] = tile[tx][ty + 8 * r];
}

// ---------------------------------------------------------------------------
// Kernel 2: dense encoder  latent = relu(x @ W_enc^T)  via bf16 WMMA.
// bf16 operands pre-staged in workspace; tiles move global->LDS with
// GLOBAL_LOAD_ASYNC_TO_LDS_B128 (ASYNCcnt), double-buffered LDS.
// ---------------------------------------------------------------------------
#define TM  128
#define TN  128
#define TK  32
#define LDK 40   // padded LDS k-stride in bf16 elems (80 B -> conflict-free b128)

__device__ __forceinline__ void wait_async0()
{
#if __has_builtin(__builtin_amdgcn_s_wait_asynccnt)
    __builtin_amdgcn_s_wait_asynccnt(0);
#else
    asm volatile("s_wait_asynccnt 0x0" ::: "memory");
#endif
}

// Issue two async b128 loads: 32 B of one tile row (both LDS and memory side
// get the same immediate offset per the CDNA5 async pseudocode).
__device__ __forceinline__ void async_row32(unsigned ldsAddr, unsigned voff,
                                            const __bf16* sbase)
{
    asm volatile(
        "global_load_async_to_lds_b128 %0, %1, %2\n\t"
        "global_load_async_to_lds_b128 %0, %1, %2 offset:16"
        :: "v"(ldsAddr), "v"(voff), "s"((uint64_t)(uintptr_t)sbase)
        : "memory");
}

__device__ __forceinline__ v16bf frag_from_lds(const __bf16* tileBase,
                                               int row, int ka, int kb)
{
    const v8bf c0 = *(const v8bf*)(tileBase + row * LDK + ka);
    const v8bf c1 = *(const v8bf*)(tileBase + row * LDK + kb);
    v16bf r;
#pragma unroll
    for (int i = 0; i < 8; ++i) { r[i] = c0[i]; r[i + 8] = c1[i]; }
    return r;
}

__global__ __launch_bounds__(256)
void sae_encoder_wmma(const __bf16* __restrict__ xb,
                      const __bf16* __restrict__ wb,
                      float* __restrict__ latent)
{
    __shared__ __bf16 As[2][TM * LDK];
    __shared__ __bf16 Bs[2][TN * LDK];

    const int t    = threadIdx.x;
    const int lane = t & 31;
    const int wave = t >> 5;
    const int wm   = wave & 1;        // M half (64 rows)
    const int wn   = wave >> 1;       // N quarter (32 cols)

    const int m0 = blockIdx.y * TM;
    const int n0 = blockIdx.x * TN;

    // staging: thread t fills 32 B (2 x b128) of row (t>>1), half (t&1)
    const int ldRow   = t >> 1;
    const int ldPairB = (t & 1) * 32;                       // byte offset in row

    const unsigned voffA = (unsigned)((m0 + ldRow) * (D_IN * 2) + ldPairB);
    const unsigned voffB = (unsigned)((n0 + ldRow) * (D_IN * 2) + ldPairB);
    unsigned ldsA[2], ldsB[2];
    ldsA[0] = (unsigned)(uintptr_t)&As[0][0] + (unsigned)(ldRow * (LDK * 2) + ldPairB);
    ldsA[1] = (unsigned)(uintptr_t)&As[1][0] + (unsigned)(ldRow * (LDK * 2) + ldPairB);
    ldsB[0] = (unsigned)(uintptr_t)&Bs[0][0] + (unsigned)(ldRow * (LDK * 2) + ldPairB);
    ldsB[1] = (unsigned)(uintptr_t)&Bs[1][0] + (unsigned)(ldRow * (LDK * 2) + ldPairB);

    // fragment addressing per CDNA5 16-bit A layout:
    // lanes 0-15: K 0..7 / 16..23 ; lanes 16-31: K 8..15 / 24..31
    const int l16 = lane & 15;
    const int ka  = (lane >> 4) * 8;
    const int kb  = 16 + (lane >> 4) * 8;

    v8f acc[4][2];
#pragma unroll
    for (int mi = 0; mi < 4; ++mi)
#pragma unroll
        for (int ni = 0; ni < 2; ++ni)
            acc[mi][ni] = (v8f)(0.0f);

    constexpr int KSTEPS = D_IN / TK;   // 24

    // prologue: fill buffer 0 for k-step 0
    async_row32(ldsA[0], voffA, xb);
    async_row32(ldsB[0], voffB, wb);

    int cur = 0;
    for (int kt = 0; kt < KSTEPS; ++kt) {
        wait_async0();          // this wave's loads into buf[cur] have landed
        __syncthreads();        // every wave's loads landed; all reads of
                                // buf[cur^1] from the previous step retired

        if (kt + 1 < KSTEPS) {  // prefetch next k-step into the other buffer
            const __bf16* sx = xb + (kt + 1) * TK;
            const __bf16* sw = wb + (kt + 1) * TK;
            async_row32(ldsA[cur ^ 1], voffA, sx);
            async_row32(ldsB[cur ^ 1], voffB, sw);
        }

        // ---- fragments + 8 WMMAs on buf[cur] ----
        const __bf16* Ab = &As[cur][0];
        const __bf16* Bb = &Bs[cur][0];
        v16bf af[4], bf[2];
#pragma unroll
        for (int mi = 0; mi < 4; ++mi)
            af[mi] = frag_from_lds(Ab, wm * 64 + mi * 16 + l16, ka, kb);
#pragma unroll
        for (int ni = 0; ni < 2; ++ni)
            bf[ni] = frag_from_lds(Bb, wn * 32 + ni * 16 + l16, ka, kb);

#pragma unroll
        for (int mi = 0; mi < 4; ++mi)
#pragma unroll
            for (int ni = 0; ni < 2; ++ni)
                acc[mi][ni] = __builtin_amdgcn_wmma_f32_16x16x32_bf16(
                    false, af[mi], false, bf[ni],
                    (short)0, acc[mi][ni], false, false);

        cur ^= 1;
    }

    // ---- epilogue: fused ReLU (single v_max_num_f32), store f32 latent ----
    // C/D layout: VGPR r, lane l -> (M = r + 8*(l>>4), N = l&15)
#pragma unroll
    for (int mi = 0; mi < 4; ++mi)
#pragma unroll
        for (int ni = 0; ni < 2; ++ni) {
            const int row = m0 + wm * 64 + mi * 16 + (lane >> 4) * 8;
            const int col = n0 + wn * 32 + ni * 16 + (lane & 15);
            float* out = latent + (size_t)row * D_HID + col;
#pragma unroll
            for (int r = 0; r < 8; ++r)
                out[(size_t)r * D_HID] = fmaxf(acc[mi][ni][r], 0.0f);
        }
}

// ---------------------------------------------------------------------------
// Kernel 3: exact per-row top-K (K=32) via 8-bit radix select.
// Post-ReLU values are >= 0, so raw IEEE bits compare correctly as u32.
// ---------------------------------------------------------------------------
__global__ __launch_bounds__(256)
void sae_topk(float* __restrict__ latent, uint2* __restrict__ pairs)
{
    __shared__ unsigned int keys[D_HID];          // 64 KB row cache
    __shared__ unsigned int hist[256];
    __shared__ unsigned int s_prefval, s_himask, s_kk, s_eq, s_pos;

    const int row = blockIdx.x;
    const int t   = threadIdx.x;
    float* lrow = latent + (size_t)row * D_HID;

#pragma unroll 4
    for (int i = 0; i < D_HID / 256; ++i)
        keys[t + 256 * i] = __float_as_uint(lrow[t + 256 * i]);

    if (t == 0) { s_prefval = 0u; s_himask = 0u; s_kk = K_TOP; s_eq = 0u; s_pos = 0u; }
    __syncthreads();

    for (int pass = 0; pass < 4; ++pass) {
        const int shift = 24 - 8 * pass;
        hist[t] = 0u;
        __syncthreads();
        const unsigned pref = s_prefval, hm = s_himask;
        for (int i = 0; i < D_HID / 256; ++i) {
            const unsigned u = keys[t + 256 * i];
            if ((u & hm) == pref)
                atomicAdd(&hist[(u >> shift) & 255u], 1u);
        }
        __syncthreads();
        if (t == 0) {
            unsigned kk = s_kk, cum = 0u; int d = 0;
            for (int b = 255; b >= 0; --b) {
                const unsigned c = hist[b];
                if (cum + c >= kk) { d = b; s_kk = kk - cum; break; }
                cum += c;
            }
            s_prefval = pref | ((unsigned)d << shift);
            s_himask  = hm   | (0xFFu << shift);
        }
        __syncthreads();
    }

    const unsigned thr     = s_prefval;  // exact K-th largest key
    const unsigned keep_eq = s_kk;       // #ties at thr to keep

    for (int i = 0; i < D_HID / 256; ++i) {
        const int h = t + 256 * i;
        const unsigned u = keys[h];
        bool keep = (u > thr);
        if (!keep && u == thr) {
            const unsigned r = atomicAdd(&s_eq, 1u);
            keep = (r < keep_eq);
        }
        lrow[h] = keep ? __uint_as_float(u) : 0.0f;
        if (keep) {
            const unsigned p = atomicAdd(&s_pos, 1u);   // exactly K kept
            pairs[(size_t)row * K_TOP + p] = make_uint2((unsigned)h, u);
        }
    }
}

// ---------------------------------------------------------------------------
// Kernel 4: sparse decoder  recon[b,:] = sum_j val_j * W_decT[h_j, :]
// ---------------------------------------------------------------------------
__global__ __launch_bounds__(256)
void sae_decode(const uint2* __restrict__ pairs,
                const float* __restrict__ WdecT,
                float* __restrict__ recon)
{
    __shared__ unsigned sh[K_TOP];
    __shared__ float    sv[K_TOP];
    const int row = blockIdx.x;
    const int t   = threadIdx.x;
    if (t < K_TOP) {
        const uint2 p = pairs[(size_t)row * K_TOP + t];
        sh[t] = p.x;
        sv[t] = __uint_as_float(p.y);
    }
    __syncthreads();

    float a0 = 0.0f, a1 = 0.0f, a2 = 0.0f;   // 768 = 3 * 256
#pragma unroll 8
    for (int j = 0; j < K_TOP; ++j) {
        const float* wr = WdecT + (size_t)sh[j] * D_IN;
        const float  v  = sv[j];
        a0 = fmaf(v, wr[t],       a0);
        a1 = fmaf(v, wr[t + 256], a1);
        a2 = fmaf(v, wr[t + 512], a2);
    }
    float* out = recon + (size_t)row * D_IN;
    out[t] = a0; out[t + 256] = a1; out[t + 512] = a2;
}

// ---------------------------------------------------------------------------
extern "C" void kernel_launch(void* const* d_in, const int* in_sizes, int n_in,
                              void* d_out, int out_size, void* d_ws, size_t ws_size,
                              hipStream_t stream)
{
    (void)in_sizes; (void)n_in; (void)out_size; (void)ws_size;

    const float* x    = (const float*)d_in[0];   // [B, D_IN]
    const float* Wenc = (const float*)d_in[1];   // [D_HID, D_IN]
    const float* Wdec = (const float*)d_in[2];   // [D_IN, D_HID]

    float* latent = (float*)d_out;                                   // [B, D_HID]
    float* recon  = (float*)d_out + (size_t)B_SZ * D_HID;            // [B, D_IN]

    // workspace layout
    char* ws = (char*)d_ws;
    float*  WdecT = (float*)ws;                              // 50.33 MB
    ws += (size_t)D_HID * D_IN * sizeof(float);
    uint2*  pairs = (uint2*)ws;                              // 2.10 MB
    ws += (size_t)B_SZ * K_TOP * sizeof(uint2);
    __bf16* xbf   = (__bf16*)ws;                             // 12.58 MB
    ws += (size_t)B_SZ * D_IN * sizeof(__bf16);
    __bf16* wbf   = (__bf16*)ws;                             // 25.17 MB

    f32_to_bf16<<<(B_SZ * D_IN) / 4 / 256, 256, 0, stream>>>(x, xbf);
    f32_to_bf16<<<(D_HID * D_IN) / 4 / 256, 256, 0, stream>>>(Wenc, wbf);
    sae_transpose<<<dim3(D_HID / 32, D_IN / 32), 256, 0, stream>>>(Wdec, WdecT);
    sae_encoder_wmma<<<dim3(D_HID / TN, B_SZ / TM), 256, 0, stream>>>(xbf, wbf, latent);
    sae_topk<<<B_SZ, 256, 0, stream>>>(latent, pairs);
    sae_decode<<<B_SZ, 256, 0, stream>>>(pairs, WdecT, recon);
}
